// SequenceLayer_40501541601288
// MI455X (gfx1250) — compile-verified, matrix operand order checked by reading, not run on hardware
//
#include <hip/hip_runtime.h>
#include <hip/hip_bf16.h>

// ---------------------------------------------------------------------------
// S4D sequence layer for MI455X (gfx1250, wave32, WMMA + TDM).
// Chunked state-space scan replaces FFT convolution:
//   per (b,h):  A_loc = W(NxT) @ Z(TxNC)          (state increments, WMMA)
//               carry  = sequential 64-step scan   (tiny, VALU)
//               y      = K(TxT)@Z + V(TxN)@carry   (WMMA)
// All matmuls f16->f32 WMMA. B operands column-major so each lane's fragment
// is two aligned 16-byte loads. z row staged into LDS via TENSOR_LOAD_TO_LDS.
// ---------------------------------------------------------------------------

#define B_  16
#define H_  256
#define L_  4096
#define N_  64      // modes
#define T_  64      // chunk length
#define NC_ 64      // number of chunks (L / T)

typedef __attribute__((ext_vector_type(16))) _Float16     v16h;
typedef __attribute__((ext_vector_type(8)))  _Float16     v8h;
typedef __attribute__((ext_vector_type(8)))  float        v8f;
typedef __attribute__((ext_vector_type(4)))  float        v4f;
typedef __attribute__((ext_vector_type(4)))  unsigned int v4u;
typedef __attribute__((ext_vector_type(8)))  int          v8i;
typedef __attribute__((ext_vector_type(4)))  int          v4i;

#if __has_builtin(__builtin_amdgcn_tensor_load_to_lds) && \
    __has_builtin(__builtin_amdgcn_s_wait_tensorcnt)
#define HAVE_TDM 1
#else
#define HAVE_TDM 0
#endif

// ---- WMMA fragment helper -------------------------------------------------
// CDNA5 f16 16x16x32 fragment: lane needs K-runs [k0+khi, +7] and
// [k0+khi+16, +7] of one row (A, row-major) / one column (B, col-major).
__device__ __forceinline__ v16h load_frag_k(const _Float16* p, int ld,
                                            int mn0, int k0, int lane) {
  int khi = (lane >> 4) * 8;
  const _Float16* base = p + (size_t)(mn0 + (lane & 15)) * ld + k0 + khi;
  v8h lo = *(const v8h*)(base);
  v8h hi = *(const v8h*)(base + 16);
  v16h f;
#pragma unroll
  for (int e = 0; e < 8; ++e) { f[e] = lo[e]; f[e + 8] = hi[e]; }
  return f;
}
__device__ __forceinline__ v8f wmma_f16(v16h a, v16h b, v8f c) {
  return __builtin_amdgcn_wmma_f32_16x16x32_f16(false, a, false, b,
                                                (short)0, c, false, false);
}

// ---------------------------------------------------------------------------
// Kernel 1: per-h precompute of chunk matrices (f16) and w^T carries (f32).
// ---------------------------------------------------------------------------
__global__ void k_precompute(const float* __restrict__ log_dt,
                             const float* __restrict__ A_re,
                             const float* __restrict__ A_im,
                             const float* __restrict__ C_re,
                             const float* __restrict__ C_im,
                             _Float16* __restrict__ Wr,  _Float16* __restrict__ Wi,
                             _Float16* __restrict__ Vr0, _Float16* __restrict__ ViN0,
                             _Float16* __restrict__ Vr1, _Float16* __restrict__ ViN1,
                             _Float16* __restrict__ K0,  _Float16* __restrict__ K1,
                             float* __restrict__ wT) {
  __shared__ float pwr[N_ * T_], pwi[N_ * T_];
  __shared__ float ctr[2][N_], cti[2][N_];
  __shared__ float ktap[2][T_];

  const int h = blockIdx.x;
  const int n = threadIdx.x;   // 64 threads: one mode each

  float dt  = __expf(log_dt[h]);
  float ar  = A_re[h * N_ + n], ai = A_im[h * N_ + n];
  float dar = dt * ar, dai = dt * ai;
  float ew  = __expf(dar);
  float wr  = ew * __cosf(dai);
  float wi  = ew * __sinf(dai);
  float den = ar * ar + ai * ai;
  if (den < 1e-20f) den = 1e-20f;
  float e1r = wr - 1.0f, e1i = wi;
  float rr = (e1r * ar + e1i * ai) / den;
  float ri = (e1i * ar - e1r * ai) / den;
#pragma unroll
  for (int d = 0; d < 2; ++d) {
    float cr = C_re[(d * H_ + h) * N_ + n];
    float ci = C_im[(d * H_ + h) * N_ + n];
    ctr[d][n] = cr * rr - ci * ri;
    cti[d][n] = cr * ri + ci * rr;
  }
  float cr_ = 1.0f, ci_ = 0.0f;            // w^0 .. w^63, leaves w^64
  for (int p = 0; p < T_; ++p) {
    pwr[n * T_ + p] = cr_; pwi[n * T_ + p] = ci_;
    float t = cr_ * wr - ci_ * wi;
    ci_ = cr_ * wi + ci_ * wr;
    cr_ = t;
  }
  wT[(h * N_ + n) * 2 + 0] = cr_;
  wT[(h * N_ + n) * 2 + 1] = ci_;

  for (int tau = 0; tau < T_; ++tau) {
    Wr[h * 4096 + n * T_ + tau] = (_Float16)pwr[n * T_ + (T_ - 1 - tau)];
    Wi[h * 4096 + n * T_ + tau] = (_Float16)pwi[n * T_ + (T_ - 1 - tau)];
  }
#pragma unroll
  for (int d = 0; d < 2; ++d) {
    _Float16* Vr  = d ? Vr1  : Vr0;
    _Float16* ViN = d ? ViN1 : ViN0;
    for (int tau = 0; tau < T_; ++tau) {
      float wpr = (tau < T_ - 1) ? pwr[n * T_ + tau + 1] : cr_;
      float wpi = (tau < T_ - 1) ? pwi[n * T_ + tau + 1] : ci_;
      float vr = ctr[d][n] * wpr - cti[d][n] * wpi;
      float vi = ctr[d][n] * wpi + cti[d][n] * wpr;
      Vr [h * 4096 + tau * N_ + n] = (_Float16)( 2.0f * vr);
      ViN[h * 4096 + tau * N_ + n] = (_Float16)(-2.0f * vi);
    }
  }
  __syncthreads();
  {
    int lag = n;
#pragma unroll
    for (int d = 0; d < 2; ++d) {
      float s = 0.0f;
      for (int m = 0; m < N_; ++m)
        s += ctr[d][m] * pwr[m * T_ + lag] - cti[d][m] * pwi[m * T_ + lag];
      ktap[d][lag] = 2.0f * s;
    }
  }
  __syncthreads();
  {
    int tau = n;
#pragma unroll
    for (int d = 0; d < 2; ++d) {
      _Float16* K = d ? K1 : K0;
      for (int j = 0; j < T_; ++j)
        K[h * 4096 + tau * T_ + j] =
            (j <= tau) ? (_Float16)ktap[d][tau - j] : (_Float16)0.0f;
    }
  }
}

// ---------------------------------------------------------------------------
// Kernel 2: LayerNorm over H for each (b,l).  Coalesced over l.
// ---------------------------------------------------------------------------
__global__ void k_layernorm(const float* __restrict__ x,
                            const float* __restrict__ ln_w,
                            const float* __restrict__ ln_b,
                            float* __restrict__ z) {
  int idx = blockIdx.x * blockDim.x + threadIdx.x;   // (b,l) flat
  int b = idx >> 12, l = idx & (L_ - 1);
  const float* xp = x + (size_t)b * H_ * L_ + l;
  float s = 0.0f, s2 = 0.0f;
  for (int h = 0; h < H_; ++h) {
    float v = xp[(size_t)h * L_];
    s += v; s2 += v * v;
  }
  float mu = s * (1.0f / H_);
  float var = s2 * (1.0f / H_) - mu * mu;
  float rs = rsqrtf(var + 1e-5f);
  float* zp = z + (size_t)b * H_ * L_ + l;
  for (int h = 0; h < H_; ++h)
    zp[(size_t)h * L_] = (xp[(size_t)h * L_] - mu) * rs * ln_w[h] + ln_b[h];
}

// ---------------------------------------------------------------------------
// Kernel 3: fused chunked bidirectional S4D per (b,h) row + D*z + GELU -> f16.
// 256 threads = 8 waves; wave w owns the tile pair (tM = w>>1,
// tN = {2(w&1), 2(w&1)+1}) so one A-fragment feeds two accumulators.
// z row staged to LDS with the Tensor Data Mover (async, TENSORcnt).
// ---------------------------------------------------------------------------
__global__ void k_s4chunk(const float* __restrict__ z,
                          const float* __restrict__ D,
                          const _Float16* __restrict__ Wr,
                          const _Float16* __restrict__ Wi,
                          const _Float16* __restrict__ Vr0,
                          const _Float16* __restrict__ ViN0,
                          const _Float16* __restrict__ Vr1,
                          const _Float16* __restrict__ ViN1,
                          const _Float16* __restrict__ K0,
                          const _Float16* __restrict__ K1,
                          const float* __restrict__ wT,
                          _Float16* __restrict__ g) {
  __shared__ __align__(16) float    zrow[L_];       // 16 KB original z row
  __shared__ __align__(16) _Float16 Zt[NC_ * T_];   //  8 KB Zt[c*64+tau]
  __shared__ __align__(16) float    Are[N_ * NC_];  // 16 KB (row-major n,c)
  __shared__ __align__(16) float    Aim[N_ * NC_];  // 16 KB
  __shared__ __align__(16) _Float16 cpr[NC_ * N_];  //  8 KB col-major [c][n]
  __shared__ __align__(16) _Float16 cpi[NC_ * N_];  //  8 KB
  __shared__ __align__(16) float    yrow[L_];       // 16 KB

  const int bh = blockIdx.x;
  const int h  = bh & (H_ - 1);
  const int tid = threadIdx.x;
  const int lane = tid & 31, wave = tid >> 5;
  const int hi = lane >> 4, lm = lane & 15;
  const int tM  = wave >> 1;
  const int tN0 = (wave & 1) * 2;

  const float* zp = z + (size_t)bh * L_;
#if HAVE_TDM
  if (wave == 0) {
    // D# for a 1-row 4096-element f32 tile: contiguous 16 KB global -> LDS.
    unsigned long long ga = (unsigned long long)(const void*)zp;
    unsigned ldsa = (unsigned)(unsigned long long)(void*)&zrow[0];
    v4u g0 = { 1u,                                   // count=1, user mode
               ldsa,                                 // lds_addr
               (unsigned)ga,                         // global_addr[31:0]
               (unsigned)((ga >> 32) & 0x01FFFFFFu) | (2u << 30) }; // +type=2
    v8i g1 = { (int)(2u << 16),        // data_size=4B
               (int)(4096u << 16),     // tensor_dim0 = 4096 (low bits)
               (int)(1u << 16),        // tensor_dim1 = 1
               (int)(4096u << 16),     // tile_dim0 = 4096
               1,                      // tile_dim1 = 1
               4096,                   // tensor_dim0_stride
               0, 0 };
    v4i gz = { 0, 0, 0, 0 };
#if defined(__clang_major__) && (__clang_major__ >= 23)
    v8i gz8 = { 0, 0, 0, 0, 0, 0, 0, 0 };
    __builtin_amdgcn_tensor_load_to_lds(g0, g1, gz, gz, gz8, 0);
#else
    __builtin_amdgcn_tensor_load_to_lds(g0, g1, gz, gz, 0);
#endif
    __builtin_amdgcn_s_wait_tensorcnt(0);
  }
  __syncthreads();
#else
  for (int i = tid; i < L_ / 4; i += 256)
    ((v4f*)zrow)[i] = ((const v4f*)zp)[i];
  __syncthreads();
#endif

  for (int dir = 0; dir < 2; ++dir) {
    // column-major Zt[c][tau] == z[c*64+tau]: identity fwd, reversed bwd
    if (dir == 0) {
      for (int i = tid; i < L_ / 8; i += 256) {
        v8h v;
#pragma unroll
        for (int e = 0; e < 8; ++e) v[e] = (_Float16)zrow[i * 8 + e];
        ((v8h*)Zt)[i] = v;
      }
    } else {
      for (int i = tid; i < L_ / 8; i += 256) {
        v8h v;
#pragma unroll
        for (int e = 0; e < 8; ++e) v[e] = (_Float16)zrow[L_ - 1 - (i * 8 + e)];
        ((v8h*)Zt)[i] = v;
      }
    }
    __syncthreads();

    // ---- A_loc = Wr@Z and Wi@Z (each 64x64x64) ----
    const _Float16* Wg[2] = {Wr + h * 4096, Wi + h * 4096};
    float* Aout[2] = {Are, Aim};
#pragma unroll
    for (int mm = 0; mm < 2; ++mm) {
      v8f acc0 = {}, acc1 = {};
#pragma unroll
      for (int k0 = 0; k0 < 64; k0 += 32) {
        v16h a = load_frag_k(Wg[mm], 64, tM * 16, k0, lane);
        acc0 = wmma_f16(a, load_frag_k(Zt, 64, tN0 * 16, k0, lane), acc0);
        acc1 = wmma_f16(a, load_frag_k(Zt, 64, (tN0 + 1) * 16, k0, lane), acc1);
      }
#pragma unroll
      for (int r = 0; r < 8; ++r) {
        int row = (tM * 16 + r + hi * 8) * NC_;
        Aout[mm][row + tN0 * 16 + lm]       = acc0[r];
        Aout[mm][row + (tN0 + 1) * 16 + lm] = acc1[r];
      }
    }
    __syncthreads();

    // ---- sequential carry scan over chunks (64 threads, one mode each) ----
    if (tid < N_) {
      int n = tid;
      float wtr = wT[(h * N_ + n) * 2 + 0];
      float wti = wT[(h * N_ + n) * 2 + 1];
      float sr = 0.0f, si = 0.0f;
      for (int c = 0; c < NC_; ++c) {
        cpr[c * N_ + n] = (_Float16)sr;   // carry BEFORE chunk c (col-major)
        cpi[c * N_ + n] = (_Float16)si;
        float nr = wtr * sr - wti * si + Are[n * NC_ + c];
        float ni = wtr * si + wti * sr + Aim[n * NC_ + c];
        sr = nr; si = ni;
      }
    }
    __syncthreads();

    // ---- y = K@Z + Vr@carry_re + (-Vi)@carry_im ----
    const _Float16* Am[3] = { (dir ? K1   : K0)   + h * 4096,
                              (dir ? Vr1  : Vr0)  + h * 4096,
                              (dir ? ViN1 : ViN0) + h * 4096 };
    const _Float16* Bm[3] = { Zt, cpr, cpi };
    v8f acc0 = {}, acc1 = {};
#pragma unroll
    for (int t = 0; t < 3; ++t) {
#pragma unroll
      for (int k0 = 0; k0 < 64; k0 += 32) {
        v16h a = load_frag_k(Am[t], 64, tM * 16, k0, lane);
        acc0 = wmma_f16(a, load_frag_k(Bm[t], 64, tN0 * 16, k0, lane), acc0);
        acc1 = wmma_f16(a, load_frag_k(Bm[t], 64, (tN0 + 1) * 16, k0, lane), acc1);
      }
    }
#pragma unroll
    for (int r = 0; r < 8; ++r) {
      int tau = tM * 16 + r + hi * 8;
      int la = (tN0 * 16 + lm) * T_ + tau;
      int lb = ((tN0 + 1) * 16 + lm) * T_ + tau;
      if (dir) {
        yrow[L_ - 1 - la] += acc0[r];
        yrow[L_ - 1 - lb] += acc1[r];
      } else {
        yrow[la] = acc0[r];
        yrow[lb] = acc1[r];
      }
    }
    __syncthreads();
  }

  // ---- epilogue: + D*z, GELU(tanh), store f16 activation ----
  float Dh = D[h];
  _Float16* gp = g + (size_t)bh * L_;
  for (int i = tid; i < L_ / 8; i += 256) {
    v8h v;
#pragma unroll
    for (int e = 0; e < 8; ++e) {
      float vv = yrow[i * 8 + e] + Dh * zrow[i * 8 + e];
      float u = 0.7978845608028654f * (vv + 0.044715f * vv * vv * vv);
      v[e] = (_Float16)(0.5f * vv * (1.0f + tanhf(u)));
    }
    ((v8h*)gp)[i] = v;
  }
}

// ---------------------------------------------------------------------------
// Kernel 4: out = x + W_out @ g + b_out.  64x64 output tiles, WMMA GEMM.
// Wt row-major (o,k); Gt column-major (l,k) so fragments are b128 loads.
// ---------------------------------------------------------------------------
__global__ void k_outproj(const float* __restrict__ x,
                          const float* __restrict__ W_out,
                          const float* __restrict__ b_out,
                          const _Float16* __restrict__ g,
                          float* __restrict__ out) {
  __shared__ __align__(16) _Float16 Wt[64 * 32];   // (o_local, k_local)
  __shared__ __align__(16) _Float16 Gt[64 * 32];   // (l_local, k_local)

  int bid = blockIdx.x;
  int b   = bid >> 8;            // 4 * 64 tiles per batch
  int rem = bid & 255;
  int o0  = (rem >> 6) * 64;
  int l0  = (rem & 63) * 64;

  const int tid = threadIdx.x;
  const int lane = tid & 31, wave = tid >> 5;
  const int hi = lane >> 4, lm = lane & 15;
  const int tM  = wave >> 1;
  const int tN0 = (wave & 1) * 2;

  v8f acc0 = {}, acc1 = {};
  for (int kk = 0; kk < H_ / 32; ++kk) {
    int kb = kk * 32;
    if (kk + 1 < H_ / 32) {   // global_prefetch_b8 of next K tile
      __builtin_prefetch(&g[((size_t)b * H_ + kb + 32 + (tid >> 3)) * L_ + l0], 0, 1);
      __builtin_prefetch(&W_out[(size_t)(o0 + (tid >> 3)) * H_ + kb + 32], 0, 1);
    }
    for (int i = tid; i < 64 * 32; i += 256) {
      int r = i >> 5, c = i & 31;
      Wt[i] = (_Float16)W_out[(size_t)(o0 + r) * H_ + kb + c];
    }
    for (int i = tid; i < 64 * 32; i += 256) {
      int r = i & 31, c = i >> 5;              // r = k_local, c = l_local
      Gt[c * 32 + r] = g[((size_t)b * H_ + kb + r) * L_ + l0 + c];
    }
    __syncthreads();
    {
      v16h a = load_frag_k(Wt, 32, tM * 16, 0, lane);
      acc0 = wmma_f16(a, load_frag_k(Gt, 32, tN0 * 16, 0, lane), acc0);
      acc1 = wmma_f16(a, load_frag_k(Gt, 32, (tN0 + 1) * 16, 0, lane), acc1);
    }
    __syncthreads();
  }
#pragma unroll
  for (int r = 0; r < 8; ++r) {
    int o = o0 + tM * 16 + r + hi * 8;
    int la = l0 + tN0 * 16 + lm;
    int lb = l0 + (tN0 + 1) * 16 + lm;
    size_t offa = ((size_t)b * H_ + o) * L_ + la;
    size_t offb = ((size_t)b * H_ + o) * L_ + lb;
    out[offa] = x[offa] + acc0[r] + b_out[o];
    out[offb] = x[offb] + acc1[r] + b_out[o];
  }
}

// ---------------------------------------------------------------------------
extern "C" void kernel_launch(void* const* d_in, const int* in_sizes, int n_in,
                              void* d_out, int out_size, void* d_ws, size_t ws_size,
                              hipStream_t stream) {
  const float* x      = (const float*)d_in[0];
  const float* log_dt = (const float*)d_in[1];
  const float* A_re   = (const float*)d_in[2];
  const float* A_im   = (const float*)d_in[3];
  const float* C_re   = (const float*)d_in[4];
  const float* C_im   = (const float*)d_in[5];
  const float* Dp     = (const float*)d_in[6];
  const float* ln_w   = (const float*)d_in[7];
  const float* ln_b   = (const float*)d_in[8];
  const float* W_out  = (const float*)d_in[9];
  const float* b_out  = (const float*)d_in[10];
  float* out = (float*)d_out;

  // workspace layout
  char* ws = (char*)d_ws;
  const size_t zBytes = (size_t)B_ * H_ * L_ * sizeof(float);      // 268 MB
  const size_t gBytes = (size_t)B_ * H_ * L_ * sizeof(_Float16);   // 134 MB
  const size_t mBytes = (size_t)H_ * 4096 * sizeof(_Float16);      //   2 MB each
  float*    z    = (float*)ws;                      ws += zBytes;
  _Float16* g    = (_Float16*)ws;                   ws += gBytes;
  _Float16* Wr   = (_Float16*)ws;                   ws += mBytes;
  _Float16* Wi   = (_Float16*)ws;                   ws += mBytes;
  _Float16* Vr0  = (_Float16*)ws;                   ws += mBytes;
  _Float16* ViN0 = (_Float16*)ws;                   ws += mBytes;
  _Float16* Vr1  = (_Float16*)ws;                   ws += mBytes;
  _Float16* ViN1 = (_Float16*)ws;                   ws += mBytes;
  _Float16* K0   = (_Float16*)ws;                   ws += mBytes;
  _Float16* K1   = (_Float16*)ws;                   ws += mBytes;
  float*    wT   = (float*)ws;                      ws += (size_t)H_ * N_ * 2 * sizeof(float);

  k_precompute<<<dim3(H_), dim3(64), 0, stream>>>(
      log_dt, A_re, A_im, C_re, C_im,
      Wr, Wi, Vr0, ViN0, Vr1, ViN1, K0, K1, wT);

  k_layernorm<<<dim3(B_ * L_ / 256), dim3(256), 0, stream>>>(x, ln_w, ln_b, z);

  k_s4chunk<<<dim3(B_ * H_), dim3(256), 0, stream>>>(
      z, Dp, Wr, Wi, Vr0, ViN0, Vr1, ViN1, K0, K1, wT, g);

  k_outproj<<<dim3(B_ * 256), dim3(256), 0, stream>>>(x, W_out, b_out, g, out);
}
